// NormAttention_29978871726343
// MI455X (gfx1250) — matrix operand order repaired and checked by reading, not compiled, not run on hardware
//
#include <hip/hip_runtime.h>

// Problem constants (fixed by the reference): B=2, H=12, L=2048, D=64.
#define BH     24
#define LSEQ   2048
#define DIM    64
#define KCHUNK 64
#define NCHUNK (LSEQ / KCHUNK)          // 32
#define WAVES  4
#define QT_PER_WAVE 2
#define QROWS_PER_BLOCK (WAVES * QT_PER_WAVE * 16)   // 128
#define NQBLK  (LSEQ / QROWS_PER_BLOCK)              // 16
#define MWORDS (LSEQ / 32)                           // 64 mask words per row

typedef __bf16 bf16_t;
typedef bf16_t v16bf __attribute__((ext_vector_type(16)));
typedef bf16_t v8bf  __attribute__((ext_vector_type(8)));
typedef bf16_t v4bf  __attribute__((ext_vector_type(4)));
typedef float  v8f   __attribute__((ext_vector_type(8)));
typedef unsigned int v4u __attribute__((ext_vector_type(4)));
typedef int v4i __attribute__((ext_vector_type(4)));
typedef int v8i __attribute__((ext_vector_type(8)));

#if __has_builtin(__builtin_amdgcn_tensor_load_to_lds)
#define USE_TDM 1
#else
#define USE_TDM 0
#endif

// ---------------- Pre-pass 1: K -> bf16 (same row-major layout) ----------------
__global__ __launch_bounds__(256)
void na_conv_k(const float* __restrict__ k, bf16_t* __restrict__ kbf)
{
    const size_t i = ((size_t)blockIdx.x * 256 + threadIdx.x) * 4;
    const float4 f = *(const float4*)&k[i];
    v4bf o; o[0] = (bf16_t)f.x; o[1] = (bf16_t)f.y; o[2] = (bf16_t)f.z; o[3] = (bf16_t)f.w;
    *(v4bf*)&kbf[i] = o;
}

// ------------- Pre-pass 2: V -> bf16 transposed: vtg[bh][d][kk] -------------
__global__ __launch_bounds__(256)
void na_conv_vt(const float* __restrict__ v, bf16_t* __restrict__ vtg)
{
    const size_t o  = ((size_t)blockIdx.x * 256 + threadIdx.x) * 4;
    const int bh  = (int)(o / (DIM * LSEQ));
    const int rem = (int)(o % (DIM * LSEQ));
    const int d   = rem / LSEQ;
    const int kk  = rem % LSEQ;          // multiple of 4
    v4bf t;
#pragma unroll
    for (int j = 0; j < 4; ++j)
        t[j] = (bf16_t)v[((size_t)bh * LSEQ + kk + j) * DIM + d];
    *(v4bf*)&vtg[o] = t;
}

// ------------- Pre-pass 3: mask -> bit-packed u32 per (row, 32-col word) -------------
__global__ __launch_bounds__(256)
void na_pack_mask(const int* __restrict__ mask, unsigned int* __restrict__ mb)
{
    const size_t g = (size_t)blockIdx.x * 256 + threadIdx.x;
    const size_t word = g >> 5;
    const int    bit  = (int)(g & 31);
    const int pred = mask[word * 32 + bit];
    const unsigned long long bal = __ballot(pred != 0);   // wave32: low 32 bits
    if (bit == 0) mb[word] = (unsigned int)bal;
}

// ---------------- TDM 2-D tile load (D# per CDNA5 ISA) ----------------
#if USE_TDM
__device__ __forceinline__
void tdm_load_2d(unsigned int lds_addr, const void* gptr,
                 unsigned int dim0, unsigned int dim1, unsigned int stride0,
                 unsigned int tile0, unsigned int tile1)
{
    const unsigned long long ga = (unsigned long long)(uintptr_t)gptr;
    v4u g0;
    g0[0] = 1u;                                            // count=1, user descriptor
    g0[1] = lds_addr;                                      // LDS byte address
    g0[2] = (unsigned int)ga;                              // global_addr[31:0]
    g0[3] = (unsigned int)((ga >> 32) & 0x1FFFFFFu) | (2u << 30);  // addr[56:32] | type=2
    v8i g1;
    g1[0] = (int)(1u << 16);                               // data_size=1 (2B), wg_mask=0
    g1[1] = (int)((dim0 & 0xFFFFu) << 16);                 // tensor_dim0[15:0]
    g1[2] = (int)(((dim0 >> 16) & 0xFFFFu) | ((dim1 & 0xFFFFu) << 16));
    g1[3] = (int)(((dim1 >> 16) & 0xFFFFu) | (tile0 << 16));
    g1[4] = (int)(tile1 & 0xFFFFu);                        // tile_dim1; tile_dim2=0
    g1[5] = (int)stride0;                                  // tensor_dim0_stride[31:0]
    g1[6] = 0;                                             // stride hi | dim1_stride lo
    g1[7] = 0;
    v4i g2; g2[0] = 0; g2[1] = 0; g2[2] = 0; g2[3] = 0;
    v4i g3 = g2;
#if __has_include(<hip/amd_detail/amd_gfx1250_TDM.h>)
    v8i g4; for (int e = 0; e < 8; ++e) g4[e] = 0;
    __builtin_amdgcn_tensor_load_to_lds(g0, g1, g2, g3, g4, 0);
#else
    __builtin_amdgcn_tensor_load_to_lds(g0, g1, g2, g3, 0);
#endif
}
#endif

// ---------------- Main fused kernel ----------------
__global__ __launch_bounds__(128)
void NormAttention_29978871726343_kernel(const bf16_t* __restrict__ kbf,
                                         const bf16_t* __restrict__ vtg,
                                         const unsigned int* __restrict__ mb,
                                         const float* __restrict__ q,
                                         const float* __restrict__ gamma,
                                         const float* __restrict__ beta,
                                         float* __restrict__ out)
{
    __shared__ bf16_t Kc[2][KCHUNK * DIM];        // [k][d] double-buffered, 2x8KB
    __shared__ bf16_t VT[2][DIM * KCHUNK];        // [d][k] double-buffered, 2x8KB
    __shared__ bf16_t Ssc[WAVES][QT_PER_WAVE][16 * KCHUNK];  // 16KB S relayout scratch

    const int tid  = threadIdx.x;
    const int wave = tid >> 5;
    const int lane = tid & 31;
    const int hl   = lane >> 4;           // half-wave 0/1
    const int lr   = lane & 15;
    const int c0   = hl * 8;

    const int bh    = blockIdx.x / NQBLK;
    const int qblk  = blockIdx.x % NQBLK;
    const int qrow0 = qblk * QROWS_PER_BLOCK + wave * (QT_PER_WAVE * 16);
    const size_t base = (size_t)bh * LSEQ * DIM;
    const bf16_t* vtbh = vtg + (size_t)bh * DIM * LSEQ;

    const unsigned int bm0 = 1u << lr;            // mask bit for k-tile t with (t&1)==0
    const unsigned int bm1 = 0x10000u << lr;      // ... and (t&1)==1

    // ---- Q tiles -> A-operand layout (row = lr; K contiguous at hl*16) ----
    v16bf Aq[QT_PER_WAVE][2];
#pragma unroll
    for (int qt = 0; qt < QT_PER_WAVE; ++qt) {
        const float* qr = q + base + (size_t)(qrow0 + qt * 16 + lr) * DIM;
#pragma unroll
        for (int e = 0; e < 8; ++e) {
            Aq[qt][0][e]     = (bf16_t)qr[2 * c0 + e];
            Aq[qt][0][e + 8] = (bf16_t)qr[2 * c0 + 8 + e];
            Aq[qt][1][e]     = (bf16_t)qr[2 * c0 + 32 + e];
            Aq[qt][1][e + 8] = (bf16_t)qr[2 * c0 + 40 + e];
        }
    }

    // O^T accumulators: per q-tile, 4 tiles of 16(d) x 16(q) f32
    v8f O[QT_PER_WAVE][4];
#pragma unroll
    for (int qt = 0; qt < QT_PER_WAVE; ++qt)
#pragma unroll
        for (int j = 0; j < 4; ++j)
#pragma unroll
            for (int e = 0; e < 8; ++e) O[qt][j][e] = 0.0f;

#if USE_TDM
    const unsigned int ldsK[2] = { (unsigned int)(uintptr_t)&Kc[0][0],
                                   (unsigned int)(uintptr_t)&Kc[1][0] };
    const unsigned int ldsV[2] = { (unsigned int)(uintptr_t)&VT[0][0],
                                   (unsigned int)(uintptr_t)&VT[1][0] };
    // Prologue: DMA chunk 0 into buffer 0 (one wave issues; EXEC-independent DMA)
    if (wave == 0) {
        tdm_load_2d(ldsK[0], kbf + base, DIM, LSEQ, DIM, DIM, KCHUNK);
        tdm_load_2d(ldsV[0], vtbh,       LSEQ, DIM, LSEQ, KCHUNK, DIM);
    }
#endif

    for (int ch = 0; ch < NCHUNK; ++ch) {
        const int buf = ch & 1;
        const int k0  = ch * KCHUNK;

#if USE_TDM
        if (wave == 0) __builtin_amdgcn_s_wait_tensorcnt(0);
        __syncthreads();                      // data ready + prev readers done
        if (wave == 0 && ch + 1 < NCHUNK) {   // prefetch next chunk via TDM
            const int kn = k0 + KCHUNK;
            tdm_load_2d(ldsK[buf ^ 1], kbf + base + (size_t)kn * DIM, DIM, LSEQ, DIM, DIM, KCHUNK);
            tdm_load_2d(ldsV[buf ^ 1], vtbh + kn,                     LSEQ, DIM, LSEQ, KCHUNK, DIM);
        }
#else
        __syncthreads();
        // Fallback cooperative stage (pure bf16 copies, no conversion)
        {
            const uint4* src = (const uint4*)(kbf + base + (size_t)k0 * DIM);
            uint4* dst = (uint4*)&Kc[buf][0];
#pragma unroll
            for (int r = 0; r < 4; ++r) dst[tid + r * 128] = src[tid + r * 128];
            const int d    = tid >> 1;
            const int half = tid & 1;
            const uint4* vs = (const uint4*)(vtbh + (size_t)d * LSEQ + k0 + half * 32);
            uint4* vd = (uint4*)&VT[buf][d * KCHUNK + half * 32];
#pragma unroll
            for (int r = 0; r < 4; ++r) vd[r] = vs[r];
        }
        __syncthreads();
#endif

        const bf16_t* KcB = Kc[buf];
        const bf16_t* VTB = VT[buf];

        // Mask words: one u64 covers a row's 64 staged k-cols
        uint2 mw[QT_PER_WAVE][8];
#pragma unroll
        for (int qt = 0; qt < QT_PER_WAVE; ++qt) {
            const int rb = qrow0 + qt * 16 + hl * 8;
#pragma unroll
            for (int i = 0; i < 8; ++i)
                mw[qt][i] = *(const uint2*)&mb[(size_t)(rb + i) * MWORDS + ch * 2];
        }

        // ---- S phase: Q x K^T for 4 k-tiles x 2 q-tiles; mask; bf16 to scratch ----
#pragma unroll
        for (int t = 0; t < 4; ++t) {
            // K B-operand (32x16 per K-block): lane holds col n=lr, K contiguous
            const bf16_t* kb = &KcB[(t * 16 + lr) * DIM + hl * 16];
            const v16bf Bk0 = *(const v16bf*)(kb);        // d = 0..31
            const v16bf Bk1 = *(const v16bf*)(kb + 32);   // d = 32..63
#pragma unroll
            for (int qt = 0; qt < QT_PER_WAVE; ++qt) {
                v8f s;
#pragma unroll
                for (int e = 0; e < 8; ++e) s[e] = 0.0f;
                s = __builtin_amdgcn_wmma_f32_16x16x32_bf16(false, Aq[qt][0], false, Bk0,
                                                            (short)0, s, false, false);
                s = __builtin_amdgcn_wmma_f32_16x16x32_bf16(false, Aq[qt][1], false, Bk1,
                                                            (short)0, s, false, false);
                const unsigned int bmt = (t & 1) ? bm1 : bm0;
#pragma unroll
                for (int i = 0; i < 8; ++i) {
                    const unsigned int w = (t & 2) ? mw[qt][i].y : mw[qt][i].x;
                    const float sv = (w & bmt) ? s[i] : -10000.0f;
                    Ssc[wave][qt][(i + hl * 8) * KCHUNK + t * 16 + lr] = (bf16_t)sv;
                }
            }
        }

        // ---- O phase: O^T += V^T x S^T ----
        v16bf BS[QT_PER_WAVE][2];
#pragma unroll
        for (int qt = 0; qt < QT_PER_WAVE; ++qt) {
            const bf16_t* sp = &Ssc[wave][qt][lr * KCHUNK + hl * 16];
            BS[qt][0] = *(const v16bf*)(sp);        // k = 0..31 block
            BS[qt][1] = *(const v16bf*)(sp + 32);   // k = 32..63 block
        }
#pragma unroll
        for (int j = 0; j < 4; ++j) {
            // V^T A-operand tiles (row d = j*16+lr), two 32-k blocks
            const bf16_t* p = &VTB[(j * 16 + lr) * KCHUNK + hl * 16];
            v16bf AV0, AV1;
            {
                v8bf a = *(const v8bf*)(p);
                v8bf b = *(const v8bf*)(p + 8);
                v8bf c = *(const v8bf*)(p + 32);
                v8bf d = *(const v8bf*)(p + 40);
#pragma unroll
                for (int e = 0; e < 8; ++e) {
                    AV0[e] = a[e]; AV0[e + 8] = b[e];
                    AV1[e] = c[e]; AV1[e + 8] = d[e];
                }
            }
#pragma unroll
            for (int qt = 0; qt < QT_PER_WAVE; ++qt) {
                O[qt][j] = __builtin_amdgcn_wmma_f32_16x16x32_bf16(false, AV0, false, BS[qt][0],
                                                                   (short)0, O[qt][j], false, false);
                O[qt][j] = __builtin_amdgcn_wmma_f32_16x16x32_bf16(false, AV1, false, BS[qt][1],
                                                                   (short)0, O[qt][j], false, false);
            }
        }
    }

    // ---- LayerNorm over D=64 (O^T layout: lane owns q=lr, d = j*16 + i + hl*8) ----
    float gv[4][8], bv[4][8];
#pragma unroll
    for (int j = 0; j < 4; ++j)
#pragma unroll
        for (int i = 0; i < 8; ++i) {
            gv[j][i] = gamma[j * 16 + c0 + i];
            bv[j][i] = beta [j * 16 + c0 + i];
        }

    const float invD = 1.0f / (float)DIM;
#pragma unroll
    for (int qt = 0; qt < QT_PER_WAVE; ++qt) {
        float s1 = 0.0f, s2 = 0.0f;
#pragma unroll
        for (int j = 0; j < 4; ++j)
#pragma unroll
            for (int e = 0; e < 8; ++e) {
                const float x = O[qt][j][e];
                s1 += x; s2 += x * x;
            }
        // row q=lr split across lanes (lr, lr+16): one xor-16 combine
        s1 += __shfl_xor(s1, 16, 32);
        s2 += __shfl_xor(s2, 16, 32);
        const float mean = s1 * invD;
        const float var  = s2 * invD - mean * mean;
        const float rstd = rsqrtf(var + 1e-12f);

        const int row = qrow0 + qt * 16 + lr;
        float* op = out + base + (size_t)row * DIM + c0;
#pragma unroll
        for (int j = 0; j < 4; ++j)
#pragma unroll
            for (int e = 0; e < 8; ++e)
                op[j * 16 + e] = (O[qt][j][e] - mean) * rstd * gv[j][e] + bv[j][e];
    }
}

extern "C" void kernel_launch(void* const* d_in, const int* in_sizes, int n_in,
                              void* d_out, int out_size, void* d_ws, size_t ws_size,
                              hipStream_t stream) {
    (void)in_sizes; (void)n_in; (void)out_size; (void)ws_size;
    const float* q     = (const float*)d_in[0];
    const float* k     = (const float*)d_in[1];
    const float* v     = (const float*)d_in[2];
    const int*   mask  = (const int*)  d_in[3];
    const float* gamma = (const float*)d_in[4];
    const float* beta  = (const float*)d_in[5];
    float* out = (float*)d_out;

    const size_t nkv = (size_t)BH * LSEQ * DIM;            // 3,145,728 elements
    bf16_t* kbf = (bf16_t*)d_ws;
    bf16_t* vtg = kbf + nkv;
    unsigned int* mb = (unsigned int*)(vtg + nkv);         // 2048*64 words

    na_conv_k   <<<(int)(nkv / 4 / 256), 256, 0, stream>>>(k, kbf);
    na_conv_vt  <<<(int)(nkv / 4 / 256), 256, 0, stream>>>(v, vtg);
    na_pack_mask<<<(LSEQ * (size_t)LSEQ) / 256, 256, 0, stream>>>(mask, mb);

    const int grid = BH * NQBLK;   // 24 * 16 = 384 blocks, 128 threads (4 waves)
    NormAttention_29978871726343_kernel<<<grid, 128, 0, stream>>>(kbf, vtg, mb, q,
                                                                  gamma, beta, out);
}
